// OpticalDepth_48189533061253
// MI455X (gfx1250) — matrix operand order repaired and checked by reading, not compiled, not run on hardware
//
#include <hip/hip_runtime.h>

typedef _Float16 v16h __attribute__((ext_vector_type(16)));
typedef _Float16 v8h  __attribute__((ext_vector_type(8)));
typedef float    v8f  __attribute__((ext_vector_type(8)));

#define N_ROWS  32768
#define TILE    128        // rows per workgroup
#define NTHR    128        // 4 waves of 32
#define H1_LD   72         // f16 stride of h1 rows (64 + pad; 144B, 16B-aligned)
#define W2T_LD  72         // f16 stride of w2t rows
#define W2T_ELT (64 * W2T_LD)       // f16 elements per net image (4608)
#define W2T_CH  576        // float4 chunks per net image (4608*2/16)
#define PL_LD   20         // f32 stride of partial-transpose rows (80B, 16B-aligned)
#define USE_ASYNC 1

// Per-gas-net species index: repeat(arange(6), [29,13,9,3,9,13])
__constant__ int d_spec[76] = {
  0,0,0,0,0,0,0,0,0,0,0,0,0,0,0,0,0,0,0,0,0,0,0,0,0,0,0,0,0,
  1,1,1,1,1,1,1,1,1,1,1,1,1,
  2,2,2,2,2,2,2,2,2,
  3,3,3,
  4,4,4,4,4,4,4,4,4,
  5,5,5,5,5,5,5,5,5,5,5,5,5
};
// First target channel per net (-1 = contributes nowhere)
__constant__ int d_chanA[76] = {
  0,1,2,3,4,5,6,-1,-1,9,10,11,12,13,14,15,16,17,18,19,20,21,22,23,24,25,26,27,28,
  0,1,2,17,18,19,20,21,22,25,26,27,28,
  0,1,2,5,6,9,10,13,14,
  0,1,2,
  0,1,2,3,4,7,8,11,12,
  0,1,2,15,16,17,18,19,20,21,22,27,28
};
// Second target channel (h2o nets 3,4 -> channels 7,8)
__constant__ int d_chanB[76] = {
  -1,-1,-1,7,8,-1,-1,-1,-1,-1,-1,-1,-1,-1,-1,-1,-1,-1,-1,-1,-1,-1,-1,-1,-1,-1,-1,-1,-1,
  -1,-1,-1,-1,-1,-1,-1,-1,-1,-1,-1,-1,-1,
  -1,-1,-1,-1,-1,-1,-1,-1,-1,
  -1,-1,-1,
  -1,-1,-1,-1,-1,-1,-1,-1,-1,
  -1,-1,-1,-1,-1,-1,-1,-1,-1,-1,-1,-1,-1
};

// One-time: W2 (76,64,64) f32 -> f16 transposed, padded LDS-image layout in d_ws.
//   image[g][n*W2T_LD + k] = (f16) W2[g][k][n]   (k >= 64 -> 0)
__global__ __launch_bounds__(NTHR)
void prep_w2t_kernel(const float* __restrict__ W2, _Float16* __restrict__ dst)
{
  const int g   = blockIdx.x;
  const int tid = threadIdx.x;
  for (int idx = tid; idx < W2T_ELT; idx += NTHR) {
    const int n = idx / W2T_LD;
    const int k = idx - n * W2T_LD;
    const float v = (k < 64) ? W2[g * 4096 + k * 64 + n] : 0.0f;
    dst[(size_t)g * W2T_ELT + idx] = (_Float16)v;
  }
}

__global__ __launch_bounds__(NTHR)
void optdepth_kernel(const float* __restrict__ t_p,
                     const float* __restrict__ comp,
                     const float* __restrict__ W1,
                     const float* __restrict__ b1,
                     const float* __restrict__ b2,
                     const float* __restrict__ Wout,
                     const float* __restrict__ bout,
                     const float* __restrict__ ke_lw,
                     const float* __restrict__ ke_iw,
                     const _Float16* __restrict__ wsrc,   // prepped W2^T images
                     float* __restrict__ out)
{
  __shared__ __align__(16) _Float16 h1[TILE * H1_LD];       // 18.0 KB
  __shared__ __align__(16) _Float16 w2t[2][W2T_ELT];        // 18.0 KB (double buffer)
  __shared__ __align__(16) float    pl[TILE * PL_LD];       // 10.0 KB partial transpose
  __shared__               float    taus[29 * TILE];        // 14.5 KB channel accumulators

  const int tid   = threadIdx.x;
  const int lane  = tid & 31;
  const int wave  = tid >> 5;
  const int lhalf = lane >> 4;
  const int l16   = lane & 15;
  const int row   = blockIdx.x * TILE + tid;

  const float x0 = t_p[2 * row];
  const float x1 = t_p[2 * row + 1];

  #pragma unroll
  for (int c = 0; c < 29; ++c) taus[c * TILE + tid] = 0.0f;

#if USE_ASYNC
  // Kick off async copy of net 0's W2^T image into buffer 0 (ASYNCcnt += 5)
  {
    const unsigned ldsbase = (unsigned)(uintptr_t)(&w2t[0][0]);
    const unsigned long long gbase = (unsigned long long)(uintptr_t)wsrc;
    #pragma unroll
    for (int i = 0; i < 5; ++i) {
      int idx = tid + i * NTHR; if (idx > W2T_CH - 1) idx = W2T_CH - 1;
      const unsigned loff = ldsbase + (unsigned)idx * 16u;
      const unsigned goff = (unsigned)idx * 16u;
      asm volatile("global_load_async_to_lds_b128 %0, %1, %2"
                   :: "v"(loff), "v"(goff), "s"(gbase) : "memory");
    }
  }
#endif

  for (int g = 0; g < 76; ++g) {
    const int buf = g & 1;
    const int gn  = (g + 1) % 76;

    // ---- Layer 1 (fp32, uniform scalar weight loads) -> h1 rows (A layout) ----
    {
      const float* w1g = &W1[g * 128];   // (2,64)
      const float* b1g = &b1[g * 64];
      #pragma unroll
      for (int h = 0; h < 64; h += 8) {
        v8h pk;
        #pragma unroll
        for (int j = 0; j < 8; ++j) {
          float a = fmaf(x0, w1g[h + j], fmaf(x1, w1g[64 + h + j], b1g[h + j]));
          a = a > 0.0f ? a : 0.0f;
          pk[j] = (_Float16)a;
        }
        *(v8h*)&h1[tid * H1_LD + h] = pk;
      }
    }

#if USE_ASYNC
    // ---- Issue async copy of NEXT net's image into the other buffer ----
    {
      const unsigned ldsbase = (unsigned)(uintptr_t)(&w2t[buf ^ 1][0]);
      const unsigned long long gbase =
          (unsigned long long)(uintptr_t)(wsrc + (size_t)gn * W2T_ELT);
      #pragma unroll
      for (int i = 0; i < 5; ++i) {
        int idx = tid + i * NTHR; if (idx > W2T_CH - 1) idx = W2T_CH - 1;
        const unsigned loff = ldsbase + (unsigned)idx * 16u;
        const unsigned goff = (unsigned)idx * 16u;
        asm volatile("global_load_async_to_lds_b128 %0, %1, %2"
                     :: "v"(loff), "v"(goff), "s"(gbase) : "memory");
      }
    }
    // Current net's 5 copies are the 5 oldest: wait until <=5 outstanding.
    asm volatile("s_wait_asynccnt 0x5" ::: "memory");
#else
    // Fallback: synchronous copy of current net's image
    {
      const float4* src = (const float4*)(wsrc + (size_t)g * W2T_ELT);
      float4* dst = (float4*)&w2t[buf][0];
      #pragma unroll
      for (int i = 0; i < 5; ++i) {
        int idx = tid + i * NTHR; if (idx > W2T_CH - 1) idx = W2T_CH - 1;
        dst[idx] = src[idx];
      }
    }
#endif
    // Warm L2 for the net after next
    __builtin_prefetch(wsrc + (size_t)((g + 2) % 76) * W2T_ELT + tid * 32, 0, 1);

    __syncthreads();   // h1 + w2t[buf] visible to all waves

    // ---- Layer 2: (128x64)@(64x64) via v_wmma_f32_16x16x32_f16, Wout folded in ----
    v16h Bf[4][2];
    #pragma unroll
    for (int ct = 0; ct < 4; ++ct) {
      #pragma unroll
      for (int kb = 0; kb < 2; ++kb) {
        const _Float16* p = &w2t[buf][(ct * 16 + l16) * W2T_LD + kb * 32 + lhalf * 8];
        union { v16h v; v8h h[2]; } u;
        u.h[0] = *(const v8h*)p;
        u.h[1] = *(const v8h*)(p + 16);
        Bf[ct][kb] = u.v;
      }
    }
    float wv[4], bv[4];
    #pragma unroll
    for (int ct = 0; ct < 4; ++ct) {
      wv[ct] = Wout[g * 64 + ct * 16 + l16];
      bv[ct] = b2[g * 64 + ct * 16 + l16];
    }

    float part[2][8];
    #pragma unroll
    for (int rt2 = 0; rt2 < 2; ++rt2) {
      const int rt = wave * 2 + rt2;
      v16h Af[2];
      #pragma unroll
      for (int kb = 0; kb < 2; ++kb) {
        const _Float16* p = &h1[(rt * 16 + l16) * H1_LD + kb * 32 + lhalf * 8];
        union { v16h v; v8h h[2]; } u;
        u.h[0] = *(const v8h*)p;          // K base .. base+7
        u.h[1] = *(const v8h*)(p + 16);   // K base+16 .. base+23
        Af[kb] = u.v;
      }
      #pragma unroll
      for (int r = 0; r < 8; ++r) part[rt2][r] = 0.0f;
      #pragma unroll
      for (int ct = 0; ct < 4; ++ct) {
        v8f c;
        #pragma unroll
        for (int r = 0; r < 8; ++r) c[r] = bv[ct];
        c = __builtin_amdgcn_wmma_f32_16x16x32_f16(false, Af[0], false, Bf[ct][0],
                                                   (short)0, c, false, false);
        c = __builtin_amdgcn_wmma_f32_16x16x32_f16(false, Af[1], false, Bf[ct][1],
                                                   (short)0, c, false, false);
        #pragma unroll
        for (int r = 0; r < 8; ++r) {
          float v = c[r] > 0.0f ? c[r] : 0.0f;   // relu(H2[m][n])
          part[rt2][r] = fmaf(v, wv[ct], part[rt2][r]);
        }
      }
    }

    // ---- Finish layer-3 dot: transpose partials through padded LDS ----
    #pragma unroll
    for (int rt2 = 0; rt2 < 2; ++rt2)
      #pragma unroll
      for (int r = 0; r < 8; ++r)
        pl[(wave * 32 + rt2 * 16 + lhalf * 8 + r) * PL_LD + l16] = part[rt2][r];
    __syncthreads();

    {
      const float4* pr = (const float4*)&pl[tid * PL_LD];
      const float4 a0 = pr[0], a1 = pr[1], a2 = pr[2], a3 = pr[3];
      float s = bout[g]
              + ((a0.x + a0.y) + (a0.z + a0.w)) + ((a1.x + a1.y) + (a1.z + a1.w))
              + ((a2.x + a2.y) + (a2.z + a2.w)) + ((a3.x + a3.y) + (a3.z + a3.w));
      const float ke  = s > 0.0f ? s : 0.0f;
      const float cm  = comp[d_spec[g] * N_ROWS + row];
      const float tau = ke * cm;
      const int ca = d_chanA[g];
      if (ca >= 0) taus[ca * TILE + tid] += tau;
      const int cb = d_chanB[g];
      if (cb >= 0) taus[cb * TILE + tid] += tau;
    }
  }

  // ---- Outputs: tau_gases (29,N), tau_lw (29,N), tau_iw (29,N), flat ----
  #pragma unroll
  for (int c = 0; c < 29; ++c)
    out[c * N_ROWS + row] = taus[c * TILE + tid];

  const float c6 = comp[6 * N_ROWS + row];
  const float c7 = comp[7 * N_ROWS + row];
  #pragma unroll
  for (int c = 0; c < 29; ++c) {
    out[(29 + c) * N_ROWS + row] = ke_lw[c] * c6;
    out[(58 + c) * N_ROWS + row] = ke_iw[c] * c7;
  }
}

extern "C" void kernel_launch(void* const* d_in, const int* in_sizes, int n_in,
                              void* d_out, int out_size, void* d_ws, size_t ws_size,
                              hipStream_t stream) {
  const float* t_p   = (const float*)d_in[0];
  const float* comp  = (const float*)d_in[1];
  const float* W1    = (const float*)d_in[2];
  const float* b1    = (const float*)d_in[3];
  const float* W2    = (const float*)d_in[4];
  const float* b2    = (const float*)d_in[5];
  const float* Wout  = (const float*)d_in[6];
  const float* bout  = (const float*)d_in[7];
  const float* ke_lw = (const float*)d_in[8];
  const float* ke_iw = (const float*)d_in[9];

  _Float16* w2t_img = (_Float16*)d_ws;   // 76 * 4608 f16 = 700 KB

  prep_w2t_kernel<<<dim3(76), dim3(NTHR), 0, stream>>>(W2, w2t_img);
  optdepth_kernel<<<dim3(N_ROWS / TILE), dim3(NTHR), 0, stream>>>(
      t_p, comp, W1, b1, b2, Wout, bout, ke_lw, ke_iw, w2t_img, (float*)d_out);
}
